// exampleLSTM_22806276341801
// MI455X (gfx1250) — matrix operand order repaired and checked by reading, not compiled, not run on hardware
//
#include <hip/hip_runtime.h>
#include <hip/hip_bf16.h>

typedef __attribute__((ext_vector_type(16))) _Float16 v16h;
typedef __attribute__((ext_vector_type(8)))  _Float16 v8h;
typedef __attribute__((ext_vector_type(8)))  float    v8f;

#define TLEN 16384
#define NB   8
#define HID  64
#define GSZ  256   // 4*H
#define FEAT 128   // input width of every layer (I == 2H == 128)

static __device__ __forceinline__ float sigm(float x) {
    return 1.0f / (1.0f + __expf(-x));
}
static __device__ __forceinline__ float fast_tanh(float x) {
    return 2.0f * sigm(2.0f * x) - 1.0f;
}

// ---------------------------------------------------------------------------
// K0: convert fp32 input sequence to f16, zero the carried h/c states.
// ---------------------------------------------------------------------------
__global__ void k_convert_init(const float* __restrict__ x, _Float16* __restrict__ seq,
                               float* __restrict__ hstate, float* __restrict__ cstate, int n)
{
    int stride = gridDim.x * blockDim.x;
    int tid = blockIdx.x * blockDim.x + threadIdx.x;
    for (int i = tid; i < n; i += stride) seq[i] = (_Float16)x[i];
    for (int i = tid; i < 2048; i += stride) { hstate[i] = 0.f; cstate[i] = 0.f; }
}

// ---------------------------------------------------------------------------
// K1: repack weights into WMMA B-fragment order (f16) and fold biases.
//  B-fragment (32x16, K x N) for v_wmma_f32_16x16x32_f16, in this code's contiguous k order (A matches):
//    lane l holds column N = l%16, K = q + 16*(l/16), q = 0..15
//  wihF: [layer][dir][nt(16)][kc(4)][lane(32)][q(16)]
//  whhF: [layer][dir][nt(16)][kc(2)][lane(32)][q(16)]
// ---------------------------------------------------------------------------
__global__ void k_prep_weights(const float* __restrict__ Wih, const float* __restrict__ Whh,
                               const float* __restrict__ bih, const float* __restrict__ bhh,
                               _Float16* __restrict__ wihF, _Float16* __restrict__ whhF,
                               float* __restrict__ biasF)
{
    int stride = gridDim.x * blockDim.x;
    int tid = blockIdx.x * blockDim.x + threadIdx.x;

    const int total_ih = 2*2*16*4*32*16;  // 131072
    for (int i = tid; i < total_ih; i += stride) {
        int q  = i & 15;
        int l  = (i >> 4) & 31;
        int kc = (i >> 9) & 3;
        int nt = (i >> 11) & 15;
        int ld = (i >> 15) & 3;           // layer*2 + dir
        int n  = nt*16 + (l & 15);
        int k  = kc*32 + q + 16*(l >> 4);
        wihF[i] = (_Float16)Wih[((size_t)ld*GSZ + n)*FEAT + k];
    }
    const int total_hh = 2*2*16*2*32*16;  // 65536
    for (int i = tid; i < total_hh; i += stride) {
        int q  = i & 15;
        int l  = (i >> 4) & 31;
        int kc = (i >> 9) & 1;
        int nt = (i >> 10) & 15;
        int ld = (i >> 14) & 3;
        int n  = nt*16 + (l & 15);
        int k  = kc*32 + q + 16*(l >> 4);
        whhF[i] = (_Float16)Whh[((size_t)ld*GSZ + n)*HID + k];
    }
    for (int i = tid; i < 1024; i += stride) biasF[i] = bih[i] + bhh[i];
}

// ---------------------------------------------------------------------------
// K2: input projection gx = seq @ Wih^T + b for one layer, both directions.
//  M = T*B = 131072 rows (M-tile = 16 rows = 2 timesteps), N = 256, K = 128.
//  Output gx is stored in WMMA C/D fragment layout:
//    gx[dir][t][nt(16)][n(16)][b(8)]  (f16)
//  grid = (256, 2): 32 M-tiles per block, blockIdx.y = dir.
// ---------------------------------------------------------------------------
__global__ void __launch_bounds__(256) k_input_proj(
    const _Float16* __restrict__ seq, const _Float16* __restrict__ wihF,
    const float* __restrict__ biasF, _Float16* __restrict__ gx, int layer)
{
    __shared__ __align__(32) _Float16 wlds[16*4*32*16];  // 64 KB

    int dir = blockIdx.y;
    const _Float16* wF   = wihF + (size_t)(layer*2 + dir) * (16*4*32*16);
    const float*    bias = biasF + (layer*2 + dir) * GSZ;

    int tid = threadIdx.x;
    for (int i = tid*8; i < 16*4*32*16; i += blockDim.x*8)
        *(v8h*)&wlds[i] = *(const v8h*)&wF[i];
    __syncthreads();

    int wave = tid >> 5, lane = tid & 31, lmod = lane & 15, lhi = lane >> 4;

    for (int r = 0; r < 4; ++r) {
        int mt = blockIdx.x * 32 + r * 8 + wave;
        // A fragment (16x32 f16): lane l row M=l%16, K = q + 16*(l/16)
        const _Float16* arow = seq + (size_t)(mt*16 + lmod) * FEAT;
        v16h a[4];
        #pragma unroll
        for (int kc = 0; kc < 4; ++kc) {
            v8h lo = *(const v8h*)&arow[kc*32 + 16*lhi];
            v8h hi = *(const v8h*)&arow[kc*32 + 16*lhi + 8];
            a[kc] = __builtin_shufflevector(lo, hi, 0,1,2,3,4,5,6,7,8,9,10,11,12,13,14,15);
        }
        int t = mt*2 + lhi;  // C/D: lanes 0-15 -> first timestep, 16-31 -> second
        _Float16* gbase = gx + ((size_t)dir*TLEN + t) * 2048;
        #pragma unroll 4
        for (int nt = 0; nt < 16; ++nt) {
            v8f acc = {};
            #pragma unroll
            for (int kc = 0; kc < 4; ++kc) {
                v16h b = *(const v16h*)&wlds[((nt*4 + kc)*32 + lane)*16];
                acc = __builtin_amdgcn_wmma_f32_16x16x32_f16(false, a[kc], false, b,
                                                             (short)0, acc, false, false);
            }
            float bv = bias[nt*16 + lmod];
            v8h oh;
            #pragma unroll
            for (int v = 0; v < 8; ++v) oh[v] = (_Float16)(acc[v] + bv);
            // lane l, vgpr v -> (b=v, n=l%16): contiguous 16B store per lane
            *(v8h*)&gbase[(nt*16 + lmod)*8] = oh;
        }
    }
}

// ---------------------------------------------------------------------------
// K3: recurrent scan, one persistent block per direction (grid = 2 x 256).
//  Per step: D[16,256] = A(h,16x64) x Whh^T + gx[t]; gates; c,h update; y[t].
//  Wave w owns N-tiles {w, w+8}: waves 0-3 -> (i,g) chunks, waves 4-7 -> (f,o).
// ---------------------------------------------------------------------------
__global__ void __launch_bounds__(256) k_scan(
    const _Float16* __restrict__ gx, const _Float16* __restrict__ whhF,
    _Float16* __restrict__ y, float* __restrict__ hstate, float* __restrict__ cstate,
    int layer)
{
    __shared__ __align__(16) _Float16 hbuf[NB * 80];   // h as f16, row pitch 80 (bank skew)
    __shared__ __align__(16) float    pbuf[4*16*8];    // sigm(i)*tanh(g)

    int dir = blockIdx.x;
    const _Float16* wF = whhF + (size_t)(layer*2 + dir) * (16*2*32*16);
    const _Float16* gb = gx + (size_t)dir * TLEN * 2048;
    float* hst = hstate + (layer*2 + dir) * 512;
    float* cst = cstate + (layer*2 + dir) * 512;

    int tid = threadIdx.x, wave = tid >> 5, lane = tid & 31, lmod = lane & 15, lhi = lane >> 4;
    int nt0 = wave, nt1 = wave + 8;
    int jb  = (wave >= 4) ? (wave - 4) : 0;

    // Whh B-fragments persistent in VGPRs
    v16h b0[2], b1[2];
    #pragma unroll
    for (int kc = 0; kc < 2; ++kc) {
        b0[kc] = *(const v16h*)&wF[((nt0*2 + kc)*32 + lane)*16];
        b1[kc] = *(const v16h*)&wF[((nt1*2 + kc)*32 + lane)*16];
    }

    // carried-in hidden/cell state
    for (int i = tid; i < 512; i += 256)
        hbuf[(i >> 6)*80 + (i & 63)] = (_Float16)hst[i];
    float c[8];
    #pragma unroll
    for (int v = 0; v < 8; ++v) c[v] = 0.f;
    if (wave >= 4 && lane < 16) {
        #pragma unroll
        for (int v = 0; v < 8; ++v) c[v] = cst[v*HID + jb*16 + lane];
    }
    __syncthreads();

    for (int s = 0; s < TLEN; ++s) {
        int t = dir ? (TLEN - 1 - s) : s;
        const _Float16* gt = gb + (size_t)t * 2048;
        if (s + 8 < TLEN) {
            int tp = dir ? (TLEN - 9 - s) : (s + 8);
            __builtin_prefetch((const void*)(gb + (size_t)tp * 2048 + tid * 8), 0, 0);
        }

        // gx fragments (real batch rows live in lanes 0-15)
        v8h g0h = {}, g1h = {};
        if (lane < 16) {
            g0h = *(const v8h*)&gt[(nt0*16 + lmod)*8];
            g1h = *(const v8h*)&gt[(nt1*16 + lmod)*8];
        }

        // build A from h in LDS; batch rows >= 8 are zero padding
        v16h a[2];
        #pragma unroll
        for (int kc = 0; kc < 2; ++kc) {
            v8h lo = {}, hi = {};
            if (lmod < 8) {
                const _Float16* hrow = &hbuf[lmod*80 + kc*32 + 16*lhi];
                lo = *(const v8h*)&hrow[0];
                hi = *(const v8h*)&hrow[8];
            }
            a[kc] = __builtin_shufflevector(lo, hi, 0,1,2,3,4,5,6,7,8,9,10,11,12,13,14,15);
        }

        v8f d0 = {}, d1 = {};
        d0 = __builtin_amdgcn_wmma_f32_16x16x32_f16(false, a[0], false, b0[0], (short)0, d0, false, false);
        d0 = __builtin_amdgcn_wmma_f32_16x16x32_f16(false, a[1], false, b0[1], (short)0, d0, false, false);
        d1 = __builtin_amdgcn_wmma_f32_16x16x32_f16(false, a[0], false, b1[0], (short)0, d1, false, false);
        d1 = __builtin_amdgcn_wmma_f32_16x16x32_f16(false, a[1], false, b1[1], (short)0, d1, false, false);

        if (wave < 4 && lane < 16) {           // d0 = i-chunk, d1 = g-chunk
            float* pb = &pbuf[(wave*16 + lane)*8];
            #pragma unroll
            for (int v = 0; v < 8; ++v)
                pb[v] = sigm(d0[v] + (float)g0h[v]) * fast_tanh(d1[v] + (float)g1h[v]);
        }
        __syncthreads();
        if (wave >= 4 && lane < 16) {          // d0 = f-chunk, d1 = o-chunk
            const float* pb = &pbuf[(jb*16 + lane)*8];
            #pragma unroll
            for (int v = 0; v < 8; ++v) {
                float ff = sigm(d0[v] + (float)g0h[v]);
                float oo = sigm(d1[v] + (float)g1h[v]);
                c[v] = ff * c[v] + pb[v];
                float hh = oo * fast_tanh(c[v]);
                _Float16 hx = (_Float16)hh;
                hbuf[v*80 + jb*16 + lane] = hx;
                y[((size_t)t*NB + v)*FEAT + dir*HID + jb*16 + lane] = hx;
            }
        }
        __syncthreads();
    }

    // carry-out state for the next chained call
    if (wave >= 4 && lane < 16) {
        #pragma unroll
        for (int v = 0; v < 8; ++v) cst[v*HID + jb*16 + lane] = c[v];
    }
    __syncthreads();
    for (int i = tid; i < 512; i += 256)
        hst[i] = (float)hbuf[(i >> 6)*80 + (i & 63)];
}

// ---------------------------------------------------------------------------
// K4: final FC + sigmoid on the last timestep. out: [8,2] f32.
// ---------------------------------------------------------------------------
__global__ void k_fc(const _Float16* __restrict__ yfin, const float* __restrict__ fcw,
                     const float* __restrict__ fcb, float* __restrict__ out)
{
    int tid = threadIdx.x;
    if (tid < 16) {
        int b = tid >> 1, k = tid & 1;
        const _Float16* row = yfin + (size_t)(TLEN - 1) * NB * FEAT + (size_t)b * FEAT;
        float s = fcb[k];
        for (int j = 0; j < FEAT; ++j) s += (float)row[j] * fcw[k*FEAT + j];
        out[b*2 + k] = sigm(s);
    }
}

// ---------------------------------------------------------------------------
extern "C" void kernel_launch(void* const* d_in, const int* in_sizes, int n_in,
                              void* d_out, int out_size, void* d_ws, size_t ws_size,
                              hipStream_t stream)
{
    (void)in_sizes; (void)n_in; (void)out_size; (void)ws_size;

    const float* x   = (const float*)d_in[0];
    const float* Wih = (const float*)d_in[1];
    const float* Whh = (const float*)d_in[2];
    const float* bih = (const float*)d_in[3];
    const float* bhh = (const float*)d_in[4];
    const float* fcw = (const float*)d_in[5];
    const float* fcb = (const float*)d_in[6];
    float* out = (float*)d_out;

    char* ws = (char*)d_ws;
    size_t off = 0;
    auto take = [&](size_t bytes) -> char* {
        char* p = ws + off;
        off = (off + bytes + 255) & ~(size_t)255;
        return p;
    };
    _Float16* seqA  = (_Float16*)take((size_t)TLEN*NB*FEAT*2);      // 32 MB
    _Float16* seqB  = (_Float16*)take((size_t)TLEN*NB*FEAT*2);      // 32 MB
    _Float16* gx    = (_Float16*)take((size_t)2*TLEN*2048*2);       // 128 MB
    _Float16* wihF  = (_Float16*)take(2*2*16*4*32*16*2);            // 256 KB
    _Float16* whhF  = (_Float16*)take(2*2*16*2*32*16*2);            // 128 KB
    float*    biasF = (float*)take(1024*4);
    float*    hstate= (float*)take(2048*4);
    float*    cstate= (float*)take(2048*4);

    k_convert_init<<<2048, 256, 0, stream>>>(x, seqA, hstate, cstate, TLEN*NB*FEAT);
    k_prep_weights<<<256, 256, 0, stream>>>(Wih, Whh, bih, bhh, wihF, whhF, biasF);

    _Float16* cur = seqA;
    _Float16* nxt = seqB;
    for (int it = 0; it < 6; ++it) {
        for (int layer = 0; layer < 2; ++layer) {
            k_input_proj<<<dim3(256, 2), 256, 0, stream>>>(cur, wihF, biasF, gx, layer);
            k_scan<<<2, 256, 0, stream>>>(gx, whhF, nxt, hstate, cstate, layer);
            _Float16* tmp = cur; cur = nxt; nxt = tmp;
        }
    }
    k_fc<<<1, 32, 0, stream>>>(cur, fcw, fcb, out);
}
